// LinearAttention_49203145343715
// MI455X (gfx1250) — matrix operand order repaired and checked by reading, not compiled, not run on hardware
//
#include <hip/hip_runtime.h>
#include <hip/hip_bf16.h>

// ---------------------------------------------------------------------------
// MI455X (gfx1250) linear-attention pipeline, bf16 WMMA everywhere.
// ---------------------------------------------------------------------------

typedef __bf16 bf16_t;
typedef __bf16 v16bf __attribute__((ext_vector_type(16)));
typedef float  v8f   __attribute__((ext_vector_type(8)));

constexpr int kH    = 8;
constexpr int kD    = 64;
constexpr int kSeq  = 8192;
constexpr int kB    = 4;
constexpr int kTok  = kB * kSeq;     // 32768
constexpr int kDim  = 256;
constexpr int kInner = kH * kD;      // 512
constexpr int kQkvN  = 3 * kInner;   // 1536
constexpr int kBH    = kB * kH;      // 32
constexpr int kCSplit = 16;          // K-split for k^T v GEMM

union FragAB { v16bf v; uint4 q[2]; };
union FragC  { v8f v; float f[8]; };

// A operand (16x32 bf16, row-major, lane m = L&15, half = L>>4):
// chunks K=[8h..8h+7] and K=[16+8h..16+8h+7]  -> two 16B loads.
__device__ __forceinline__ v16bf load_a_frag(const bf16_t* row, int k0, int half) {
  FragAB f;
  f.q[0] = *reinterpret_cast<const uint4*>(row + k0 + 8 * half);
  f.q[1] = *reinterpret_cast<const uint4*>(row + k0 + 8 * half + 16);
  return f.v;
}

// B operand (32x16 bf16), stored as B^T (column contiguous over K):
// lane n = L&15, half = L>>4: K=[8h..8h+7] and K=[16+8h..16+8h+7] -> two 16B loads.
__device__ __forceinline__ v16bf load_b_frag(const bf16_t* col, int k0, int half) {
  FragAB f;
  const bf16_t* p = col + k0 + 8 * half;
  f.q[0] = *reinterpret_cast<const uint4*>(p);
  f.q[1] = *reinterpret_cast<const uint4*>(p + 16);
  return f.v;
}

__device__ __forceinline__ v8f wmma_bf16(v16bf a, v16bf b, v8f c) {
  return __builtin_amdgcn_wmma_f32_16x16x32_bf16(false, a, false, b, (short)0, c,
                                                 false, false);
}

#define ZERO8 {0.f,0.f,0.f,0.f,0.f,0.f,0.f,0.f}

// ---------------------------------------------------------------------------
// Conversion / init kernels
// ---------------------------------------------------------------------------
__global__ void k_cvt_x(const float* __restrict__ x, bf16_t* __restrict__ xb, int n) {
  int i = blockIdx.x * 256 + threadIdx.x;
  if (i < n) xb[i] = (bf16_t)x[i];
}

__global__ void k_cvt_wqkv(const float* __restrict__ w, bf16_t* __restrict__ wt) {
  int i = blockIdx.x * 256 + threadIdx.x;            // wt[o][k] = w[k][o]
  if (i < kQkvN * kDim) {
    int o = i / kDim, k = i % kDim;
    wt[i] = (bf16_t)w[(size_t)k * kQkvN + o];
  }
}

__global__ void k_cvt_wout(const float* __restrict__ w, bf16_t* __restrict__ wt) {
  int i = blockIdx.x * 256 + threadIdx.x;            // wt[o][k] = w[k][o]
  if (i < kDim * kInner) {
    int o = i / kInner, k = i % kInner;
    wt[i] = (bf16_t)w[(size_t)k * kDim + o];
  }
}

__global__ void k_zero_f32(float* __restrict__ p, int n) {
  int i = blockIdx.x * 256 + threadIdx.x;
  if (i < n) p[i] = 0.f;
}

__global__ void k_cvt_g(const float* __restrict__ G, bf16_t* __restrict__ Gb, int n) {
  int i = blockIdx.x * 256 + threadIdx.x;
  if (i < n) Gb[i] = (bf16_t)(G[i] * (1.0f / (float)kSeq));
}

// ---------------------------------------------------------------------------
// GEMM1: qkv = x @ W_qkv  (M=32768, N=1536, K=256), scatter to per-head bufs
// grid (TOK/64, QKVN/32), 256 threads = 8 waves (4 M-tiles x 2 N-tiles)
// ---------------------------------------------------------------------------
__global__ __launch_bounds__(256) void k_gemm_qkv(
    const bf16_t* __restrict__ xb, const bf16_t* __restrict__ wt,
    bf16_t* __restrict__ qh, bf16_t* __restrict__ kh, bf16_t* __restrict__ vh) {
  int lane = threadIdx.x & 31, wave = threadIdx.x >> 5;
  int half = lane >> 4, l16 = lane & 15;
  int tm = blockIdx.x * 64 + (wave >> 1) * 16;
  int tn = blockIdx.y * 32 + (wave & 1) * 16;

  const bf16_t* ap = xb + (size_t)(tm + l16) * kDim;
  const bf16_t* bp = wt + (size_t)(tn + l16) * kDim;

  v8f c = ZERO8;
#pragma unroll
  for (int k0 = 0; k0 < kDim; k0 += 32) {
    v16bf a = load_a_frag(ap, k0, half);
    v16bf b = load_b_frag(bp, k0, half);
    c = wmma_bf16(a, b, c);
  }

  int which = tn / kInner;           // 0=q 1=k 2=v  (16-wide tiles never straddle)
  int h  = (tn % kInner) / kD;
  int d0 = tn % kD;
  bf16_t* dst = (which == 0) ? qh : ((which == 1) ? kh : vh);
  int b = tm / kSeq;                 // 16-wide tiles never straddle batch
  int col = d0 + l16;

  FragC fc; fc.v = c;
#pragma unroll
  for (int r = 0; r < 8; r++) {
    int tok = tm + half * 8 + r;
    int nl = tok - b * kSeq;
    dst[(((size_t)b * kH + h) * kSeq + nl) * kD + col] = (bf16_t)fc.f[r];
  }
}

// ---------------------------------------------------------------------------
// Fused instance-norm (k,v) + 2D rotary (q,k), transposed bf16 outputs.
// grid (SEQ/32, B*H), 256 threads. LDS-staged 32 tokens x (q,k,v) rows.
// Outputs: qb [bh][n][d] (A-layout for q@dots), ktb/vtb [bh][d][n] (K-major).
// ---------------------------------------------------------------------------
__global__ __launch_bounds__(256) void k_norm_rope(
    const bf16_t* __restrict__ qh, const bf16_t* __restrict__ kh,
    const bf16_t* __restrict__ vh, const float* __restrict__ pos,
    bf16_t* __restrict__ qb, bf16_t* __restrict__ ktb, bf16_t* __restrict__ vtb) {
  __shared__ float sh[32 * 192];          // [token][0..63 q | 64..127 k | 128..191 v]
  __shared__ float s_mean[2][32];
  __shared__ float s_rstd[2][32];

  int bh = blockIdx.y;
  int b  = bh >> 3;
  int n0 = blockIdx.x * 32;
  int tid = threadIdx.x;

  size_t base = ((size_t)bh * kSeq + n0) * kD;
  for (int i = tid; i < 32 * 64; i += 256) {
    int t = i >> 6, d = i & 63;
    sh[t * 192 + d]        = (float)qh[base + (size_t)t * kD + d];
    sh[t * 192 + 64 + d]   = (float)kh[base + (size_t)t * kD + d];
    sh[t * 192 + 128 + d]  = (float)vh[base + (size_t)t * kD + d];
  }
  __syncthreads();

  if (tid < 64) {                          // 32 tokens x {k,v} stats
    int t = tid >> 1, sel = tid & 1;
    const float* p = &sh[t * 192 + 64 + sel * 64];
    float s = 0.f, ss = 0.f;
    for (int d = 0; d < 64; d++) { float x = p[d]; s += x; ss += x * x; }
    float mean = s * (1.f / 64.f);
    float var  = ss * (1.f / 64.f) - mean * mean;
    s_mean[sel][t] = mean;
    s_rstd[sel][t] = rsqrtf(var + 1e-5f);
  }
  __syncthreads();

  int lane = tid & 31, w = tid >> 5;       // token = lane, d-group = wave
  int t = lane;
  int n = n0 + t;
  float px = pos[((size_t)b * kSeq + n) * 2 + 0];
  float py = pos[((size_t)b * kSeq + n) * 2 + 1];
  float km = s_mean[0][t], kr = s_rstd[0][t];
  float vm = s_mean[1][t], vr = s_rstd[1][t];
  const float* rowq = &sh[t * 192];
  const float* rowk = rowq + 64;
  const float* rowv = rowk + 64;

#pragma unroll
  for (int j = 0; j < 8; j++) {
    int d = w * 8 + j;
    int hx = d >> 5;                        // 0: x-rotary half, 1: y-rotary half
    int i  = d & 31;                        // index within rotary half
    int ip = (i ^ 16) + hx * 32;            // rotate_half partner index
    float tpos = (hx ? py : px) * 64.0f;    // SCALE / MIN_FREQ = 64
    // inv_freq[j] = 10000^(-j/16) = exp(-j * ln(10000)/16)
    float F = tpos * __expf(-(float)(i & 15) * 0.57564627324851148f);
    float sn, cs;
    __sincosf(F, &sn, &cs);
    float sgn = (i < 16) ? -1.f : 1.f;

    float qv = rowq[d] * cs + sgn * rowq[ip] * sn;
    float kvn  = (rowk[d]  - km) * kr;
    float kvnp = (rowk[ip] - km) * kr;
    float kv = kvn * cs + sgn * kvnp * sn;
    float vv = (rowv[d] - vm) * vr;

    qb[((size_t)bh * kSeq + n) * kD + d]        = (bf16_t)qv;
    ktb[((size_t)bh * kD + d) * kSeq + n]       = (bf16_t)kv;  // coalesced over n
    vtb[((size_t)bh * kD + d) * kSeq + n]       = (bf16_t)vv;
  }
}

// ---------------------------------------------------------------------------
// G[e][d] = sum_n v'[n][e] * k'[n][d]   (A = vtb rows, B^T = ktb rows)
// grid (kCSplit, B*H), 512 threads = 16 waves -> 4x4 tiles of 16x16 over 64x64.
// K-split partial sums merged with global f32 atomics.
// ---------------------------------------------------------------------------
__global__ __launch_bounds__(512) void k_kv_outer(
    const bf16_t* __restrict__ vtb, const bf16_t* __restrict__ ktb,
    float* __restrict__ G) {
  int bh = blockIdx.y;
  int kbase = blockIdx.x * (kSeq / kCSplit);
  int lane = threadIdx.x & 31, wave = threadIdx.x >> 5;
  int half = lane >> 4, l16 = lane & 15;
  int eT = (wave >> 2) * 16;
  int dT = (wave & 3) * 16;

  const bf16_t* ap = vtb + ((size_t)bh * kD + eT + l16) * kSeq + kbase;
  const bf16_t* bp = ktb + ((size_t)bh * kD + dT + l16) * kSeq + kbase;

  v8f c = ZERO8;
#pragma unroll 4
  for (int k0 = 0; k0 < kSeq / kCSplit; k0 += 32) {
    v16bf a = load_a_frag(ap, k0, half);
    v16bf b = load_b_frag(bp, k0, half);
    c = wmma_bf16(a, b, c);
  }

  FragC fc; fc.v = c;
  float* g = G + (size_t)bh * kD * kD;
  int col = dT + l16;
#pragma unroll
  for (int r = 0; r < 8; r++) {
    int e = eT + half * 8 + r;
    atomicAdd(&g[(size_t)e * kD + col], fc.f[r]);
  }
}

// ---------------------------------------------------------------------------
// H[b][n][h*64+e] = sum_d q'[bh][n][d] * dots[d][e],  dots[d][e] = Gb[e][d]
// grid ((SEQ/64)*2, B*H), 256 threads = 8 waves (4 M-tiles x 2 N-tiles).
// ---------------------------------------------------------------------------
__global__ __launch_bounds__(256) void k_qdots(
    const bf16_t* __restrict__ qb, const bf16_t* __restrict__ Gb,
    bf16_t* __restrict__ H) {
  int bh = blockIdx.y;
  int b = bh >> 3, h = bh & 7;
  int mb = blockIdx.x >> 1;
  int nb = blockIdx.x & 1;
  int lane = threadIdx.x & 31, wave = threadIdx.x >> 5;
  int half = lane >> 4, l16 = lane & 15;
  int mT = mb * 64 + (wave >> 1) * 16;
  int nT = nb * 32 + (wave & 1) * 16;

  const bf16_t* ap = qb + ((size_t)bh * kSeq + mT + l16) * kD;
  const bf16_t* bp = Gb + ((size_t)bh * kD + nT + l16) * kD;

  v8f c = ZERO8;
#pragma unroll
  for (int k0 = 0; k0 < kD; k0 += 32) {
    v16bf a = load_a_frag(ap, k0, half);
    v16bf b2 = load_b_frag(bp, k0, half);
    c = wmma_bf16(a, b2, c);
  }

  FragC fc; fc.v = c;
  int col = nT + l16;
#pragma unroll
  for (int r = 0; r < 8; r++) {
    int tok = mT + half * 8 + r;
    H[((size_t)b * kSeq + tok) * kInner + h * kD + col] = (bf16_t)fc.f[r];
  }
}

// ---------------------------------------------------------------------------
// out = H @ W_out + b_out  (M=32768, N=256, K=512)
// grid (TOK/64, DIM/32), 256 threads = 8 waves.
// ---------------------------------------------------------------------------
__global__ __launch_bounds__(256) void k_out_proj(
    const bf16_t* __restrict__ H, const bf16_t* __restrict__ wot,
    const float* __restrict__ bout, float* __restrict__ out) {
  int lane = threadIdx.x & 31, wave = threadIdx.x >> 5;
  int half = lane >> 4, l16 = lane & 15;
  int mT = blockIdx.x * 64 + (wave >> 1) * 16;
  int nT = blockIdx.y * 32 + (wave & 1) * 16;

  const bf16_t* ap = H + (size_t)(mT + l16) * kInner;
  const bf16_t* bp = wot + (size_t)(nT + l16) * kInner;

  v8f c = ZERO8;
#pragma unroll 8
  for (int k0 = 0; k0 < kInner; k0 += 32) {
    v16bf a = load_a_frag(ap, k0, half);
    v16bf b = load_b_frag(bp, k0, half);
    c = wmma_bf16(a, b, c);
  }

  FragC fc; fc.v = c;
  int col = nT + l16;
  float bias = bout[col];
#pragma unroll
  for (int r = 0; r < 8; r++) {
    int tok = mT + half * 8 + r;
    out[(size_t)tok * kDim + col] = fc.f[r] + bias;
  }
}

// ---------------------------------------------------------------------------
// Host launcher
// ---------------------------------------------------------------------------
extern "C" void kernel_launch(void* const* d_in, const int* in_sizes, int n_in,
                              void* d_out, int out_size, void* d_ws, size_t ws_size,
                              hipStream_t stream) {
  const float* x    = (const float*)d_in[0];
  const float* pos  = (const float*)d_in[1];
  const float* Wqkv = (const float*)d_in[2];
  const float* Wout = (const float*)d_in[3];
  const float* bout = (const float*)d_in[4];
  float* out = (float*)d_out;

  char* ws = (char*)d_ws;
  size_t off = 0;
  auto take = [&](size_t bytes) -> char* {
    char* p = ws + off;
    off = (off + bytes + 255) & ~(size_t)255;
    return p;
  };

  bf16_t* xb    = (bf16_t*)take((size_t)kTok * kDim * 2);          // 16 MB
  bf16_t* wqkvt = (bf16_t*)take((size_t)kQkvN * kDim * 2);         // 768 KB
  bf16_t* wot   = (bf16_t*)take((size_t)kDim * kInner * 2);        // 256 KB
  bf16_t* qh    = (bf16_t*)take((size_t)kBH * kSeq * kD * 2);      // 32 MB
  bf16_t* kh    = (bf16_t*)take((size_t)kBH * kSeq * kD * 2);      // 32 MB
  bf16_t* vh    = (bf16_t*)take((size_t)kBH * kSeq * kD * 2);      // 32 MB
  bf16_t* qb    = (bf16_t*)take((size_t)kBH * kSeq * kD * 2);      // 32 MB
  bf16_t* ktb   = (bf16_t*)take((size_t)kBH * kSeq * kD * 2);      // 32 MB
  bf16_t* vtb   = (bf16_t*)take((size_t)kBH * kSeq * kD * 2);      // 32 MB
  float*  G     = (float*)take((size_t)kBH * kD * kD * 4);         // 512 KB
  bf16_t* Gb    = (bf16_t*)take((size_t)kBH * kD * kD * 2);        // 256 KB
  bf16_t* Hb    = qh;  // qh dead after k_norm_rope; reuse (same size)

  // 0) conversions + accumulator reset (re-run each call: deterministic)
  k_cvt_x<<<(kTok * kDim + 255) / 256, 256, 0, stream>>>(x, xb, kTok * kDim);
  k_cvt_wqkv<<<(kQkvN * kDim + 255) / 256, 256, 0, stream>>>(Wqkv, wqkvt);
  k_cvt_wout<<<(kDim * kInner + 255) / 256, 256, 0, stream>>>(Wout, wot);
  k_zero_f32<<<(kBH * kD * kD + 255) / 256, 256, 0, stream>>>(G, kBH * kD * kD);

  // 1) qkv projection (WMMA)
  k_gemm_qkv<<<dim3(kTok / 64, kQkvN / 32), 256, 0, stream>>>(xb, wqkvt, qh, kh, vh);

  // 2) instance-norm + 2D rotary, transposed bf16 outputs
  k_norm_rope<<<dim3(kSeq / 32, kBH), 256, 0, stream>>>(qh, kh, vh, pos, qb, ktb, vtb);

  // 3) dots^T = v'^T k'  (WMMA, K-split + f32 atomics)
  k_kv_outer<<<dim3(kCSplit, kBH), 512, 0, stream>>>(vtb, ktb, G);

  // 4) scale by 1/N and convert to bf16
  k_cvt_g<<<(kBH * kD * kD + 255) / 256, 256, 0, stream>>>(G, Gb, kBH * kD * kD);

  // 5) q @ dots  (WMMA)
  k_qdots<<<dim3((kSeq / 64) * 2, kBH), 256, 0, stream>>>(qb, Gb, Hb);

  // 6) output projection + bias (WMMA)
  k_out_proj<<<dim3(kTok / 64, kDim / 32), 256, 0, stream>>>(Hb, wot, bout, out);
}
